// VQVAEZGrouped_19035295056274
// MI455X (gfx1250) — matrix operand order repaired and checked by reading, not compiled, not run on hardware
//
#include <hip/hip_runtime.h>

typedef __attribute__((ext_vector_type(16))) _Float16 v16h;
typedef __attribute__((ext_vector_type(8)))  float    v8f;

#define Bn 16
#define Cc 256
#define Gg 8
#define CG 32
#define Kc 1024
#define HW 1024                      // h*w = 32*32
#define NPOS (Bn * HW)               // 16384 spatial positions
#define ENC_ELEMS (Bn * Cc * HW)     // 4194304
#define ZIDX_ELEMS (Bn * Gg * HW)    // 131072
#define OUT_ZOFF  ENC_ELEMS
#define OUT_QOFF  (ENC_ELEMS + ZIDX_ELEMS)

// ---------------------------------------------------------------------------
// Prep: codebooks f32 -> f16 (WMMA operand precision) + per-code squared norms
// ---------------------------------------------------------------------------
__global__ void vq_precompute(const float* __restrict__ cb,
                              _Float16* __restrict__ cbh,
                              float* __restrict__ cnorm) {
    int idx = blockIdx.x * blockDim.x + threadIdx.x;   // code row: 0 .. 8191
    if (idx >= Gg * Kc) return;
    const float* row = cb + idx * CG;
    _Float16* orow   = cbh + idx * CG;
    float s = 0.f;
#pragma unroll
    for (int i = 0; i < CG; i++) {
        float v = row[i];
        s += v * v;
        orow[i] = (_Float16)v;
    }
    cnorm[idx] = s;
}

// ---------------------------------------------------------------------------
// Main: one wave32 per (16-position tile, group). WMMA 16x16x32 cross-term,
// running argmin in D-matrix layout, fused encode_out copy, gathered quant.
// ---------------------------------------------------------------------------
__global__ void __launch_bounds__(32) vq_main(const float* __restrict__ x,
                                              const float* __restrict__ cb,
                                              const _Float16* __restrict__ cbh,
                                              const float* __restrict__ cnorm,
                                              float* __restrict__ out) {
    const int g    = blockIdx.x & (Gg - 1);
    const int tile = blockIdx.x >> 3;
    const int n0   = tile * 16;          // 16 consecutive positions, same b
    const int b    = n0 >> 10;
    const int hw0  = n0 & (HW - 1);
    const int lane = threadIdx.x;        // 0..31
    const int half = lane >> 4;
    const int m16  = lane & 15;

    // ---- A fragment: 16 positions x 32 dims, f16 (contiguous K = 16*half + e, same order as B)
    //      lanes 0-15: row=lane, K = {0..15}; lanes 16-31: K = {16..31}.
    //      Fuse encode_out = x while the data is in registers.
    const int hw = hw0 + m16;
    const int xbase = (b * Cc + g * CG) * HW + hw;     // + cc*HW per element
    v16h a;
#pragma unroll
    for (int e = 0; e < 16; e++) {
        int K = e + 16 * half;
        float v = x[xbase + K * HW];
        out[xbase + K * HW] = v;                       // encode_out
        a[e] = (_Float16)v;
    }

    float bestv[8];
    int   besti[8];
#pragma unroll
    for (int r = 0; r < 8; r++) { bestv[r] = 3.4e38f; besti[r] = 0; }

    const _Float16* cbg = cbh + g * Kc * CG;
    const float*    cng = cnorm + g * Kc;

    // ---- 64 code tiles of 16: score = ||c||^2 - 2 * (e . c)
    for (int kk = 0; kk < Kc / 16; kk++) {
        const int k0    = kk * 16;
        const int codeN = k0 + m16;                    // this lane's code column
        const v16h bfrag = *(const v16h*)(cbg + codeN * CG + half * 16);
        const float cn   = cng[codeN];
        if (kk + 1 < Kc / 16)
            __builtin_prefetch(cbg + (codeN + 16) * CG + half * 16, 0, 0);

        v8f acc = {};
        acc = __builtin_amdgcn_wmma_f32_16x16x32_f16(
            /*neg_a=*/false, a, /*neg_b=*/false, bfrag,
            /*c_mod=*/(short)0, acc, /*reuse_a=*/false, /*reuse_b=*/false);

#pragma unroll
        for (int r = 0; r < 8; r++) {                  // row = r + 8*half
            float s = cn - 2.0f * acc[r];
            if (s < bestv[r]) { bestv[r] = s; besti[r] = codeN; }
        }
    }

    // ---- reduce the 16 lanes sharing each row (xor masks stay within halves)
#pragma unroll
    for (int mask = 1; mask < 16; mask <<= 1) {
#pragma unroll
        for (int r = 0; r < 8; r++) {
            float ov = __shfl_xor(bestv[r], mask, 32);
            int   oi = __shfl_xor(besti[r], mask, 32);
            if (ov < bestv[r] || (ov == bestv[r] && oi < besti[r])) {
                bestv[r] = ov; besti[r] = oi;
            }
        }
    }

    __shared__ int zs[16];
    if (lane == 0) {
#pragma unroll
        for (int r = 0; r < 8; r++) zs[r] = besti[r];      // rows 0..7
    }
    if (lane == 16) {
#pragma unroll
        for (int r = 0; r < 8; r++) zs[8 + r] = besti[r];  // rows 8..15
    }
    __syncthreads();

    // ---- zidx (int32), layout (b, g, h, w): 16 contiguous coalesced stores
    int* zout = (int*)out + OUT_ZOFF + (b * Gg + g) * HW + hw0;
    if (lane < 16) zout[lane] = zs[lane];

    // ---- quant_out (b, c, h, w): gather winning code rows (L2-resident f32
    //      codebook) and scatter; each half-wave writes one 64B segment.
    float* qbase = out + OUT_QOFF + (b * Cc + g * CG) * HW + hw0;
    const int zi = zs[m16];
    const float* cbrow = cb + (g * Kc + zi) * CG;
#pragma unroll
    for (int p = 0; p < 16; p++) {
        int cc = 2 * p + half;
        qbase[cc * HW + m16] = cbrow[cc];
    }
}

extern "C" void kernel_launch(void* const* d_in, const int* in_sizes, int n_in,
                              void* d_out, int out_size, void* d_ws, size_t ws_size,
                              hipStream_t stream) {
    const float* x  = (const float*)d_in[0];      // (16,256,32,32) f32
    const float* cb = (const float*)d_in[1];      // (8,1024,32)    f32

    _Float16* cbh  = (_Float16*)d_ws;                                   // 512 KB
    float*    cnrm = (float*)((char*)d_ws + (size_t)Gg * Kc * CG * 2);  // +32 KB
    float*    out  = (float*)d_out;

    vq_precompute<<<(Gg * Kc + 255) / 256, 256, 0, stream>>>(cb, cbh, cnrm);
    vq_main<<<(NPOS / 16) * Gg, 32, 0, stream>>>(x, cb, cbh, cnrm, out);
}
